// CTrans2_19928648253730
// MI455X (gfx1250) — matrix operand order repaired and checked by reading, not compiled, not run on hardware
//
#include <hip/hip_runtime.h>
#include <hip/hip_bf16.h>
#include <stdint.h>

// ---------------------------------------------------------------------------
// Problem dims (fixed by reference): D=8, C=512, H=W=64 -> N = 4096
// ---------------------------------------------------------------------------
#define DD 8
#define CC 512
#define NN 4096

typedef __attribute__((ext_vector_type(16))) __bf16    v16bf;
typedef __attribute__((ext_vector_type(8)))  float     v8f;
typedef __attribute__((ext_vector_type(4)))  unsigned  u32x4;

union Frag { u32x4 u[2]; v16bf v; };

// ---------------------------------------------------------------------------
// GEMM tiling: C[m][n] = sum_k A[m][k] * B[n][k]   ("NT" gemm, both row-major,
// K contiguous).  Block tile 128x128, 256 threads = 8 waves arranged 4(M)x2(N),
// each wave computes 32x64 via 2x4 v_wmma_f32_16x16x32_bf16 accumulators.
// LDS rows padded to 20 dwords (16 data + 4) -> conflict-free b128 frag loads.
// Triple-buffered LDS fed by GLOBAL_LOAD_ASYNC_TO_LDS_B128 (ASYNCcnt),
// prefetch depth 2: stage s computes while s+1 and s+2 are in flight.
// ---------------------------------------------------------------------------
#define BM   128
#define BN   128
#define BK   32
#define LDST 20                 // dwords per LDS row
#define BUFDW (BM * LDST)       // dwords per buffer per tile
#define NBUF 3

// 16-byte async copy: global -> LDS, tracked by ASYNCcnt (CDNA5 path).
__device__ __forceinline__ void async_copy16(const uint32_t* g, unsigned* l)
{
    asm volatile("global_load_async_to_lds_b128 %0, %1, off"
                 :
                 : "v"((unsigned)(uintptr_t)l), "v"((uint64_t)(uintptr_t)g)
                 : "memory");
}

// Async loads complete in order: waiting to <=4 guarantees the oldest stage
// (4 chunks/thread) has landed while the next stage's 4 remain in flight.
__device__ __forceinline__ void wait_async0()
{
    asm volatile("s_wait_asynccnt 0x0" ::: "memory");
}
__device__ __forceinline__ void wait_async4()
{
    asm volatile("s_wait_asynccnt 0x4" ::: "memory");
}

// Issue one K-stage of async copies: 128 rows x 16 dwords per tile,
// 512 u32x4 chunks per tile over 256 threads -> 2 A + 2 B chunks per thread.
__device__ __forceinline__ void stage_issue(
    const uint32_t* __restrict__ Ag, const uint32_t* __restrict__ Bg,
    int ldaDW, int ldbDW, int kdw,
    unsigned* la, unsigned* lb, int tid)
{
#pragma unroll
    for (int it = 0; it < 2; ++it) {
        int c   = tid + it * 256;
        int row = c >> 2;
        int q4  = (c & 3) << 2;
        async_copy16(Ag + (size_t)row * ldaDW + kdw + q4, la + row * LDST + q4);
        async_copy16(Bg + (size_t)row * ldbDW + kdw + q4, lb + row * LDST + q4);
    }
}

template <int K>
__device__ __forceinline__ void gemm_core(
    const __bf16* __restrict__ Asrc,   // pre-offset to M-tile base
    const __bf16* __restrict__ Bsrc,   // pre-offset to N-tile base
    int ldaDW, int ldbDW,
    unsigned* ldsA, unsigned* ldsB,    // each NBUF*BUFDW dwords
    v8f acc[2][4])
{
    const int tid  = threadIdx.x;      // 0..255
    const int lane = tid & 31;
    const int wave = tid >> 5;         // 0..7
    const int wm   = wave >> 1;        // 0..3 : M sub-tile (32 rows)
    const int wn   = wave & 1;         // 0..1 : N sub-tile (64 cols)
    const int half = lane >> 4;        // 0/1  : lane half (K striping)
    const int r16  = lane & 15;

    const v8f zero = {0.f,0.f,0.f,0.f,0.f,0.f,0.f,0.f};
#pragma unroll
    for (int i = 0; i < 2; ++i)
#pragma unroll
        for (int j = 0; j < 4; ++j) acc[i][j] = zero;

    const uint32_t* Ag = (const uint32_t*)Asrc;
    const uint32_t* Bg = (const uint32_t*)Bsrc;

    constexpr int nStages = K / BK;    // >= 16 for both gemms

    // Prologue: stages 0 and 1 into buffers 0 and 1.
    stage_issue(Ag, Bg, ldaDW, ldbDW, 0,        ldsA,         ldsB,         tid);
    stage_issue(Ag, Bg, ldaDW, ldbDW, BK / 2,   ldsA + BUFDW, ldsB + BUFDW, tid);

    int bufc = 0;                      // buffer holding stage s
    for (int s = 0; s < nStages; ++s) {
        if (s + 1 < nStages) wait_async4();   // stage s landed; s+1 in flight
        else                 wait_async0();
        __syncthreads();               // all waves' stage-s data visible;
                                       // all reads of the reissue target done

        if (s + 2 < nStages) {
            int bn = bufc + 2; if (bn >= NBUF) bn -= NBUF;
            stage_issue(Ag, Bg, ldaDW, ldbDW, (s + 2) * (BK / 2),
                        ldsA + bn * BUFDW, ldsB + bn * BUFDW, tid);
        }

        const unsigned* la = ldsA + bufc * BUFDW;
        const unsigned* lb = ldsB + bufc * BUFDW;

        // A fragment (16x32 bf16): dwords [half*4 .. +3] and [8+half*4 .. +3]
        Frag af[2];
#pragma unroll
        for (int i = 0; i < 2; ++i) {
            const unsigned* rp = la + (wm * 32 + i * 16 + r16) * LDST;
            af[i].u[0] = *(const u32x4*)(rp + half * 4);
            af[i].u[1] = *(const u32x4*)(rp + 8 + half * 4);
        }
        // B fragment (32x16 bf16): lane = column, dwords [half*4 .. +3] and [8+half*4 .. +3]
        Frag bfr[4];
#pragma unroll
        for (int j = 0; j < 4; ++j) {
            const unsigned* rp = lb + (wn * 64 + j * 16 + r16) * LDST;
            bfr[j].u[0] = *(const u32x4*)(rp + half * 4);
            bfr[j].u[1] = *(const u32x4*)(rp + 8 + half * 4);
        }
#pragma unroll
        for (int i = 0; i < 2; ++i)
#pragma unroll
            for (int j = 0; j < 4; ++j)
                acc[i][j] = __builtin_amdgcn_wmma_f32_16x16x32_bf16(
                    false, af[i].v, false, bfr[j].v,
                    (short)0, acc[i][j], false, false);

        bufc = (bufc + 1 >= NBUF) ? 0 : bufc + 1;
    }
}

// ---------------------------------------------------------------------------
// Kernel 1: x (f32, [d][c][n]) -> q bf16 [d][c][n] and qt bf16 [d][n][c]
// 32x32 LDS tiled transpose, block (32,8)
// ---------------------------------------------------------------------------
__global__ __launch_bounds__(256) void convert_kernel(
    const float* __restrict__ x, __bf16* __restrict__ q, __bf16* __restrict__ qt)
{
    __shared__ float tile[32][33];
    const int n0 = blockIdx.x * 32;
    const int c0 = blockIdx.y * 32;
    const int d  = blockIdx.z;
    const float*  xd = x  + (size_t)d * CC * NN;
    __bf16*       qd = q  + (size_t)d * CC * NN;
    __bf16*      qtd = qt + (size_t)d * NN * CC;

#pragma unroll
    for (int i = 0; i < 32; i += 8) {
        float v = xd[(size_t)(c0 + threadIdx.y + i) * NN + n0 + threadIdx.x];
        tile[threadIdx.y + i][threadIdx.x] = v;
        qd[(size_t)(c0 + threadIdx.y + i) * NN + n0 + threadIdx.x] = (__bf16)v;
    }
    __syncthreads();
#pragma unroll
    for (int i = 0; i < 32; i += 8) {
        qtd[(size_t)(n0 + threadIdx.y + i) * CC + c0 + threadIdx.x] =
            (__bf16)tile[threadIdx.x][threadIdx.y + i];
    }
}

// ---------------------------------------------------------------------------
// Kernel 2: energy[d][i][j] = sum_n q[d][i][n] * q[d][j][n]
// ---------------------------------------------------------------------------
__global__ __launch_bounds__(256) void gemm1_kernel(
    const __bf16* __restrict__ q, float* __restrict__ energy)
{
    __shared__ unsigned ldsA[NBUF * BUFDW];
    __shared__ unsigned ldsB[NBUF * BUFDW];
    const int d = blockIdx.z;
    const __bf16* Asrc = q + (size_t)d * CC * NN + (size_t)blockIdx.y * BM * NN;
    const __bf16* Bsrc = q + (size_t)d * CC * NN + (size_t)blockIdx.x * BN * NN;
    float* C = energy + (size_t)d * CC * CC + (size_t)blockIdx.y * BM * CC
                      + (size_t)blockIdx.x * BN;

    v8f acc[2][4];
    gemm_core<NN>(Asrc, Bsrc, NN / 2, NN / 2, ldsA, ldsB, acc);

    const int lane = threadIdx.x & 31;
    const int wave = threadIdx.x >> 5;
    const int wm = wave >> 1, wn = wave & 1;
    const int half = lane >> 4, r16 = lane & 15;
#pragma unroll
    for (int i = 0; i < 2; ++i) {
        const int rbase = wm * 32 + i * 16 + half * 8;
#pragma unroll
        for (int v = 0; v < 8; ++v) {
            float* cp = C + (size_t)(rbase + v) * CC + wn * 64 + r16;
#pragma unroll
            for (int j = 0; j < 4; ++j) cp[j * 16] = acc[i][j][v];
        }
    }
}

// ---------------------------------------------------------------------------
// Kernel 3: double softmax over rows; writes attention TRANSPOSED as bf16:
//   attT[d][j][i] = softmax_j( softmax_j(rowmax - energy)[i][j] + atten[d][i][j] )
// One 512-thread block per (d, i) row.
// ---------------------------------------------------------------------------
__device__ __forceinline__ float block_reduce(float v, float* red, int tid, bool isMax)
{
    red[tid] = v;
    __syncthreads();
#pragma unroll
    for (int s = 256; s > 0; s >>= 1) {
        if (tid < s) {
            float a = red[tid], b = red[tid + s];
            red[tid] = isMax ? fmaxf(a, b) : (a + b);
        }
        __syncthreads();
    }
    float r = red[0];
    __syncthreads();
    return r;
}

__global__ __launch_bounds__(512) void softmax_kernel(
    const float* __restrict__ energy, const float* __restrict__ atten,
    __bf16* __restrict__ attT)
{
    __shared__ float red[512];
    const int row = blockIdx.x;        // d*512 + i
    const int d = row >> 9;
    const int i = row & 511;
    const int j = threadIdx.x;

    float e = energy[(size_t)row * CC + j];
    float rowmax = block_reduce(e, red, j, true);
    float en = rowmax - e;                       // sign-flipped centering
    float m1 = block_reduce(en, red, j, true);
    float ex = __expf(en - m1);
    float s1 = block_reduce(ex, red, j, false);
    float a1 = ex / s1;

    float z  = a1 + atten[(size_t)d * CC * CC + (size_t)i * CC + j];
    float m2 = block_reduce(z, red, j, true);
    float e2 = __expf(z - m2);
    float s2 = block_reduce(e2, red, j, false);

    attT[(size_t)d * CC * CC + (size_t)j * CC + i] = (__bf16)(e2 / s2);
}

// ---------------------------------------------------------------------------
// Kernel 4: out[d][j][n] = gamma * sum_i att[d][i][j] * q[d][i][n] + x[d][j][n]
//   = NT gemm: A = attT [j][i], B = qt [n][i], K = i
// ---------------------------------------------------------------------------
__global__ __launch_bounds__(256) void gemm2_kernel(
    const __bf16* __restrict__ attT, const __bf16* __restrict__ qt,
    const float* __restrict__ x, const float* __restrict__ gamma,
    float* __restrict__ out)
{
    __shared__ unsigned ldsA[NBUF * BUFDW];
    __shared__ unsigned ldsB[NBUF * BUFDW];
    const int d = blockIdx.z;
    const __bf16* Asrc = attT + (size_t)d * CC * CC + (size_t)blockIdx.y * BM * CC;
    const __bf16* Bsrc = qt   + (size_t)d * NN * CC + (size_t)blockIdx.x * BN * CC;
    const size_t cOff = (size_t)d * CC * NN + (size_t)blockIdx.y * BM * NN
                      + (size_t)blockIdx.x * BN;

    v8f acc[2][4];
    gemm_core<CC>(Asrc, Bsrc, CC / 2, CC / 2, ldsA, ldsB, acc);

    const float g = gamma[0];
    const int lane = threadIdx.x & 31;
    const int wave = threadIdx.x >> 5;
    const int wm = wave >> 1, wn = wave & 1;
    const int half = lane >> 4, r16 = lane & 15;
#pragma unroll
    for (int i = 0; i < 2; ++i) {
        const int rbase = wm * 32 + i * 16 + half * 8;
#pragma unroll
        for (int v = 0; v < 8; ++v) {
            const float* xp = x   + cOff + (size_t)(rbase + v) * NN + wn * 64 + r16;
            float*       op = out + cOff + (size_t)(rbase + v) * NN + wn * 64 + r16;
#pragma unroll
            for (int j = 0; j < 4; ++j)
                op[j * 16] = g * acc[i][j][v] + xp[j * 16];
        }
    }
}

// ---------------------------------------------------------------------------
// Launch
// ---------------------------------------------------------------------------
extern "C" void kernel_launch(void* const* d_in, const int* in_sizes, int n_in,
                              void* d_out, int out_size, void* d_ws, size_t ws_size,
                              hipStream_t stream)
{
    const float* x     = (const float*)d_in[0];   // [8,512,64,64]
    const float* atten = (const float*)d_in[1];   // [8,512,512]
    const float* gamma = (const float*)d_in[2];   // [1]
    float*       out   = (float*)d_out;           // [8,512,64,64]

    char* ws = (char*)d_ws;
    __bf16* q      = (__bf16*)(ws);                          // 33,554,432 B
    __bf16* qt     = (__bf16*)(ws + 33554432);               // 33,554,432 B
    float*  energy = (float*) (ws + 67108864);               //  8,388,608 B
    __bf16* attT   = (__bf16*)(ws + 75497472);               //  4,194,304 B

    // 1) fp32 -> bf16 (plain + transposed)
    convert_kernel<<<dim3(NN / 32, CC / 32, DD), dim3(32, 8), 0, stream>>>(x, q, qt);
    // 2) energy = q . q^T   (K = 4096)
    gemm1_kernel<<<dim3(CC / BN, CC / BM, DD), dim3(256), 0, stream>>>(q, energy);
    // 3) double softmax -> attT (bf16, transposed)
    softmax_kernel<<<dim3(DD * CC), dim3(512), 0, stream>>>(energy, atten, attT);
    // 4) out = gamma * (att^T . q) + x   (K = 512)
    gemm2_kernel<<<dim3(NN / BN, CC / BM, DD), dim3(256), 0, stream>>>(
        attT, qt, x, gamma, out);
}
